// Mix_2834678415610
// MI455X (gfx1250) — matrix operand order repaired and checked by reading, not compiled, not run on hardware
//
#include <hip/hip_runtime.h>
#include <cstdint>
#include <cstddef>

// ---------------------------------------------------------------------------
// MI455X / gfx1250 GCN-forward implementation.
// Dense GEMMs run on v_wmma_f32_16x16x32_bf16 (wave32 WMMA, f32 accum).
// Weights are pre-packed into a B-fragment layout with contiguous k order (k = 16*half + j); each block
// stages its 4-N-tile B panel in LDS once (ds_load_b128 in the hot loop),
// and each wave keeps 4 accumulators so one A fragment feeds 4 WMMAs.
// ---------------------------------------------------------------------------

#define GN 50000
#define GE 800000
#define GF0 512
#define GH 256
#define GC 64

typedef __attribute__((ext_vector_type(16))) __bf16 v16bf;
typedef __attribute__((ext_vector_type(8)))  float  v8f;
typedef __attribute__((ext_vector_type(8)))  unsigned short us8;

__device__ __forceinline__ unsigned short f32_to_bf16(float f) {
  unsigned int u = __float_as_uint(f);
  u += 0x7FFFu + ((u >> 16) & 1u);   // round-to-nearest-even
  return (unsigned short)(u >> 16);
}

// ---- elementwise f32 -> bf16 convert (grid-stride) -------------------------
__global__ __launch_bounds__(256) void k_cvt_bf16(const float* __restrict__ in,
                                                  unsigned short* __restrict__ out,
                                                  long long n) {
  long long i = (long long)blockIdx.x * blockDim.x + threadIdx.x;
  long long stride = (long long)gridDim.x * blockDim.x;
  for (; i < n; i += stride) out[i] = f32_to_bf16(in[i]);
}

// ---- pack fp32 KxN weight matrix into wave32 WMMA B-fragment order ---------
// out[((nt*ktiles + kt)*32 + lane)*16 + j] =
//     bf16(B[(kt*32 + (lane>>4)*16 + j)*NC + nt*16 + (lane&15)])
__global__ __launch_bounds__(256) void k_pack_b(const float* __restrict__ B,
                                                unsigned short* __restrict__ out,
                                                int K, int NC) {
  int tid = blockIdx.x * blockDim.x + threadIdx.x;
  int total = K * NC;
  if (tid >= total) return;
  int ktiles = K >> 5;
  int j    = tid & 15;
  int lane = (tid >> 4) & 31;
  int kt   = (tid >> 9) % ktiles;
  int nt   = tid / (ktiles << 9);
  int col  = nt * 16 + (lane & 15);
  int k    = kt * 32 + ((lane >> 4) << 4) + j;
  out[tid] = f32_to_bf16(B[(size_t)k * NC + col]);
}

// ---- WMMA GEMM: C[M x NC] = act(A[M x K](bf16) @ Bpacked + bias) -----------
// Block = 8 waves, each wave owns one 16-row M tile and FOUR 16-col N tiles
// (blockIdx.y selects the group of 4). The 4-tile B panel (<=64KB) is staged
// in LDS once per block; A fragments are loaded once per k-step and reused
// across the 4 WMMAs.
#define MAX_KTILES 16                          // K <= 512
__global__ __launch_bounds__(256)
void k_gemm_bf16(const unsigned short* __restrict__ A,
                 const unsigned short* __restrict__ Bp,
                 float* __restrict__ C,
                 const float* __restrict__ bias,
                 int M, int K, int NC, int relu) {
  __shared__ unsigned short Bs[4 * MAX_KTILES * 512];  // [nt][kt][lane][16] = 64KB

  const int lane  = threadIdx.x & 31;
  const int wave  = threadIdx.x >> 5;
  const int ktiles = K >> 5;
  const int nt0   = blockIdx.y * 4;

  // ---- cooperative stage of the contiguous 4-tile B panel into LDS --------
  {
    const us8* __restrict__ src = (const us8*)(Bp + (size_t)nt0 * ktiles * 512);
    us8* dst = (us8*)Bs;
    const int totalv = ktiles * 256;           // (4*ktiles*512)/8 us8 elements
    for (int i = threadIdx.x; i < totalv; i += 256) dst[i] = src[i];
  }
  __syncthreads();

  const int mtile = blockIdx.x * 8 + wave;
  if (mtile * 16 >= M) return;

  const int m0   = mtile * 16;
  const int row  = m0 + (lane & 15);
  const int koff = (lane >> 4) << 4;           // A k-offset within 32-chunk
  const unsigned short* __restrict__ arow = A + (size_t)row * K;

  v8f acc[4];
#pragma unroll
  for (int n = 0; n < 4; ++n) acc[n] = (v8f){0.f,0.f,0.f,0.f,0.f,0.f,0.f,0.f};

  for (int kt = 0; kt < ktiles; ++kt) {
    union { v16bf v; us8 u[2]; } af;
    const int kb = (kt << 5) + koff;
    af.u[0] = *(const us8*)(arow + kb);        // K = koff .. koff+7
    af.u[1] = *(const us8*)(arow + kb + 8);    // K = koff+8 .. koff+15
#pragma unroll
    for (int n = 0; n < 4; ++n) {
      union { v16bf v; us8 u[2]; } bf;
      const unsigned short* bp = Bs + ((size_t)(n * ktiles + kt) * 32 + lane) * 16;
      bf.u[0] = *(const us8*)(bp);             // ds_load_b128
      bf.u[1] = *(const us8*)(bp + 8);
      acc[n] = __builtin_amdgcn_wmma_f32_16x16x32_bf16(
          /*neg_a=*/false, af.v, /*neg_b=*/false, bf.v,
          /*c_mod=*/(short)0, acc[n], /*reuse_a=*/false, /*reuse_b=*/false);
    }
  }

  const int rbase = m0 + ((lane >> 4) << 3);
#pragma unroll
  for (int n = 0; n < 4; ++n) {
    const int col = (nt0 + n) * 16 + (lane & 15);
    const float bv = bias ? bias[col] : 0.0f;
#pragma unroll
    for (int r = 0; r < 8; ++r) {
      float v = acc[n][r] + bv;
      if (relu) v = fmaxf(v, 0.0f);
      C[(size_t)(rbase + r) * NC + col] = v;
    }
  }
}

// ---- SpMM: out[dst[e]] += w[e] * h[src[e]]  (one edge per wave, H=256) -----
__global__ __launch_bounds__(256)
void k_spmm(const int* __restrict__ src, const int* __restrict__ dst,
            const float* __restrict__ ew, const float* __restrict__ h,
            float* __restrict__ out, int E) {
  const int e = blockIdx.x * 8 + (threadIdx.x >> 5);
  if (e >= E) return;
  const int lane = threadIdx.x & 31;
  const int s = src[e];
  const int d = dst[e];
  const float w = ew[e];
  const float4* __restrict__ hs = (const float4*)(h + (size_t)s * GH);
  float* __restrict__ od = out + (size_t)d * GH;
  __builtin_prefetch(od, 1, 1);
#pragma unroll
  for (int half = 0; half < 2; ++half) {
    const int q = half * 32 + lane;            // float4 index, 64 per row
    float4 v = hs[q];
    float* p = od + q * 4;
    __hip_atomic_fetch_add(p + 0, w * v.x, __ATOMIC_RELAXED, __HIP_MEMORY_SCOPE_AGENT);
    __hip_atomic_fetch_add(p + 1, w * v.y, __ATOMIC_RELAXED, __HIP_MEMORY_SCOPE_AGENT);
    __hip_atomic_fetch_add(p + 2, w * v.z, __ATOMIC_RELAXED, __HIP_MEMORY_SCOPE_AGENT);
    __hip_atomic_fetch_add(p + 3, w * v.w, __ATOMIC_RELAXED, __HIP_MEMORY_SCOPE_AGENT);
  }
}

// ---- bias (+optional relu) then f32 -> bf16 over N x 256 -------------------
__global__ __launch_bounds__(256)
void k_bias_cvt(const float* __restrict__ acc, const float* __restrict__ bias,
                unsigned short* __restrict__ out, long long total, int relu) {
  long long i = (long long)blockIdx.x * blockDim.x + threadIdx.x;
  long long stride = (long long)gridDim.x * blockDim.x;
  for (; i < total; i += stride) {
    float v = acc[i] + bias[(int)(i & (GH - 1))];
    if (relu) v = fmaxf(v, 0.0f);
    out[i] = f32_to_bf16(v);
  }
}

// ---- log_softmax over 256 cols, write bf16 into concat buffer (ld=512) -----
__global__ __launch_bounds__(256)
void k_lsm256_bf16(const float* __restrict__ Y, unsigned short* __restrict__ out,
                   int coff, int M) {
  const int row = blockIdx.x * 8 + (threadIdx.x >> 5);
  if (row >= M) return;
  const int lane = threadIdx.x & 31;
  const float* __restrict__ y = Y + (size_t)row * GH;
  float v[8];
  float mx = -__builtin_inff();
#pragma unroll
  for (int i = 0; i < 8; ++i) { v[i] = y[i * 32 + lane]; mx = fmaxf(mx, v[i]); }
#pragma unroll
  for (int off = 16; off > 0; off >>= 1) mx = fmaxf(mx, __shfl_xor(mx, off, 32));
  float s = 0.0f;
#pragma unroll
  for (int i = 0; i < 8; ++i) s += __expf(v[i] - mx);
#pragma unroll
  for (int off = 16; off > 0; off >>= 1) s += __shfl_xor(s, off, 32);
  const float lse = mx + __logf(s);
  unsigned short* __restrict__ o = out + (size_t)row * (2 * GH) + coff;
#pragma unroll
  for (int i = 0; i < 8; ++i) o[i * 32 + lane] = f32_to_bf16(v[i] - lse);
}

// ---- log_softmax over 64 cols, in place on d_out ---------------------------
__global__ __launch_bounds__(256)
void k_lsm64_inplace(float* __restrict__ Y, int M) {
  const int row = blockIdx.x * 8 + (threadIdx.x >> 5);
  if (row >= M) return;
  const int lane = threadIdx.x & 31;
  float* __restrict__ y = Y + (size_t)row * GC;
  float v0 = y[lane], v1 = y[32 + lane];
  float mx = fmaxf(v0, v1);
#pragma unroll
  for (int off = 16; off > 0; off >>= 1) mx = fmaxf(mx, __shfl_xor(mx, off, 32));
  float s = __expf(v0 - mx) + __expf(v1 - mx);
#pragma unroll
  for (int off = 16; off > 0; off >>= 1) s += __shfl_xor(s, off, 32);
  const float lse = mx + __logf(s);
  y[lane] = v0 - lse;
  y[32 + lane] = v1 - lse;
}

// ---------------------------------------------------------------------------
static inline size_t alignup(size_t x) { return (x + 255) & ~(size_t)255; }

extern "C" void kernel_launch(void* const* d_in, const int* in_sizes, int n_in,
                              void* d_out, int out_size, void* d_ws, size_t ws_size,
                              hipStream_t stream) {
  (void)in_sizes; (void)n_in; (void)out_size;

  const float* x[2]   = {(const float*)d_in[0],  (const float*)d_in[1]};
  const int*   esrc   = (const int*)d_in[2];
  const int*   edst   = (const int*)d_in[3];
  const float* ew     = (const float*)d_in[4];
  const float* W1[2]  = {(const float*)d_in[5],  (const float*)d_in[11]};
  const float* b1[2]  = {(const float*)d_in[6],  (const float*)d_in[12]};
  const float* W2[2]  = {(const float*)d_in[7],  (const float*)d_in[13]};
  const float* b2[2]  = {(const float*)d_in[8],  (const float*)d_in[14]};
  const float* LWl[2] = {(const float*)d_in[9],  (const float*)d_in[15]};
  const float* Lbl[2] = {(const float*)d_in[10], (const float*)d_in[16]};
  const float* LW     = (const float*)d_in[17];
  const float* Lb     = (const float*)d_in[18];
  float* out          = (float*)d_out;

  // ---- workspace carve-up --------------------------------------------------
  const size_t sz_xb   = (size_t)GN * GF0 * 2;      // bf16 input, per branch
  const size_t sz_g    = (size_t)GN * GH * 4;       // GEMM f32 output
  const size_t sz_acc  = (size_t)GN * GH * 4;       // SpMM f32 accumulator
  const size_t sz_hb   = (size_t)GN * GH * 2;       // bf16 hidden
  const size_t sz_cat  = (size_t)GN * (2 * GH) * 2; // bf16 concat [h1|h2]
  const size_t sz_w1p  = (size_t)GF0 * GH * 2;
  const size_t sz_w2p  = (size_t)GH * GH * 2;
  const size_t sz_lwlp = (size_t)GH * GH * 2;
  const size_t sz_lwp  = (size_t)(2 * GH) * GC * 2;

  size_t off = 0;
  char* ws = (char*)d_ws;
  unsigned short* xb   = (unsigned short*)(ws + off); off += alignup(sz_xb);
  float*          g    = (float*)         (ws + off); off += alignup(sz_g);
  float*          acc  = (float*)         (ws + off); off += alignup(sz_acc);
  unsigned short* hb   = (unsigned short*)(ws + off); off += alignup(sz_hb);
  unsigned short* cat  = (unsigned short*)(ws + off); off += alignup(sz_cat);
  unsigned short* w1p  = (unsigned short*)(ws + off); off += alignup(sz_w1p);
  unsigned short* w2p  = (unsigned short*)(ws + off); off += alignup(sz_w2p);
  unsigned short* lwlp = (unsigned short*)(ws + off); off += alignup(sz_lwlp);
  unsigned short* lwp  = (unsigned short*)(ws + off); off += alignup(sz_lwp);
  if (off > ws_size) return;  // workspace too small; bail without OOB

  const dim3 blk(256);
  const int mtiles = GN / 16;                  // 3125
  const int gx     = (mtiles + 7) / 8;         // 391 blocks, 8 waves each
  const dim3 g_gemmH(gx, GH / 64);             // NC=256 -> 4 n-groups of 4 tiles
  const dim3 g_gemmC(gx, GC / 64);             // NC=64  -> 1 n-group of 4 tiles
  const int g_spmm = (GE + 7) / 8;             // one edge per wave
  const int g_rows = (GN + 7) / 8;             // one row per wave

  // pack final classifier weight once
  hipLaunchKernelGGL(k_pack_b, dim3(((2 * GH) * GC + 255) / 256), blk, 0, stream,
                     LW, lwp, 2 * GH, GC);

  for (int br = 0; br < 2; ++br) {
    // pack branch weights into WMMA B-fragment order
    hipLaunchKernelGGL(k_pack_b, dim3((GF0 * GH + 255) / 256), blk, 0, stream,
                       W1[br], w1p, GF0, GH);
    hipLaunchKernelGGL(k_pack_b, dim3((GH * GH + 255) / 256), blk, 0, stream,
                       W2[br], w2p, GH, GH);
    hipLaunchKernelGGL(k_pack_b, dim3((GH * GH + 255) / 256), blk, 0, stream,
                       LWl[br], lwlp, GH, GH);

    // x -> bf16
    hipLaunchKernelGGL(k_cvt_bf16, dim3(4096), blk, 0, stream,
                       x[br], xb, (long long)GN * GF0);

    // g = x @ W1
    hipLaunchKernelGGL(k_gemm_bf16, g_gemmH, blk, 0, stream,
                       xb, w1p, g, (const float*)nullptr, GN, GF0, GH, 0);
    // acc = spmm(adj, g)
    hipMemsetAsync(acc, 0, sz_acc, stream);
    hipLaunchKernelGGL(k_spmm, dim3(g_spmm), blk, 0, stream,
                       esrc, edst, ew, g, acc, GE);
    // hb = bf16(relu(acc + b1))
    hipLaunchKernelGGL(k_bias_cvt, dim3(4096), blk, 0, stream,
                       acc, b1[br], hb, (long long)GN * GH, 1);

    // g = h @ W2
    hipLaunchKernelGGL(k_gemm_bf16, g_gemmH, blk, 0, stream,
                       hb, w2p, g, (const float*)nullptr, GN, GH, GH, 0);
    // acc = spmm(adj, g)
    hipMemsetAsync(acc, 0, sz_acc, stream);
    hipLaunchKernelGGL(k_spmm, dim3(g_spmm), blk, 0, stream,
                       esrc, edst, ew, g, acc, GE);
    // hb = bf16(acc + b2)
    hipLaunchKernelGGL(k_bias_cvt, dim3(4096), blk, 0, stream,
                       acc, b2[br], hb, (long long)GN * GH, 0);

    // g = h @ LWl + Lbl
    hipLaunchKernelGGL(k_gemm_bf16, g_gemmH, blk, 0, stream,
                       hb, lwlp, g, Lbl[br], GN, GH, GH, 0);
    // cat[:, br*256 : br*256+256] = bf16(log_softmax(g))
    hipLaunchKernelGGL(k_lsm256_bf16, dim3(g_rows), blk, 0, stream,
                       g, cat, br * GH, GN);
  }

  // out = cat @ LW + Lb ; log_softmax in place
  hipLaunchKernelGGL(k_gemm_bf16, g_gemmC, blk, 0, stream,
                     cat, lwp, out, Lb, GN, 2 * GH, GC, 0);
  hipLaunchKernelGGL(k_lsm64_inplace, dim3(g_rows), blk, 0, stream, out, GN);
}
